// PesudoLabelGenerator_v1_8_ot_rev_o_3702261809693
// MI455X (gfx1250) — compile-verified, matrix operand order checked by reading, not run on hardware
//
#include <hip/hip_runtime.h>
#include <hip/hip_bf16.h>

typedef __attribute__((ext_vector_type(16))) __bf16 v16bf;
typedef __attribute__((ext_vector_type(8)))  __bf16 v8bf;
typedef __attribute__((ext_vector_type(8)))  float  v8f;
typedef __attribute__((ext_vector_type(2)))  float  v2f;

#define CFEAT 128
#define OUTC  20
#define RECMAX 1e8f   // 1/EPS : min(rw1*rw2, 1e8) == 1/max(w1*w2, 1e-8)

__device__ __forceinline__ unsigned short f32_to_bf16_rne(float f) {
  unsigned int u = __float_as_uint(f);
  unsigned int lsb = (u >> 16) & 1u;
  u += 0x7fffu + lsb;
  return (unsigned short)(u >> 16);
}

// ---------------- pose: diff = inv(posses[1]) @ posses[0], single thread ----
__global__ void pose_kernel(const float* __restrict__ posses, float* __restrict__ diffM) {
  if (threadIdx.x != 0 || blockIdx.x != 0) return;
  float a[4][8];
  for (int i = 0; i < 4; ++i)
    for (int j = 0; j < 4; ++j) { a[i][j] = posses[16 + i * 4 + j]; a[i][j + 4] = (i == j) ? 1.f : 0.f; }
  for (int c = 0; c < 4; ++c) {
    int p = c; float mx = fabsf(a[c][c]);
    for (int r = c + 1; r < 4; ++r) { float v = fabsf(a[r][c]); if (v > mx) { mx = v; p = r; } }
    if (p != c) for (int j = 0; j < 8; ++j) { float t = a[c][j]; a[c][j] = a[p][j]; a[p][j] = t; }
    float inv = 1.0f / a[c][c];
    for (int j = 0; j < 8; ++j) a[c][j] *= inv;
    for (int r = 0; r < 4; ++r) if (r != c) {
      float f = a[r][c];
      for (int j = 0; j < 8; ++j) a[r][j] -= f * a[c][j];
    }
  }
  for (int i = 0; i < 4; ++i)
    for (int j = 0; j < 4; ++j) {
      float s = 0.f;
      for (int k = 0; k < 4; ++k) s += a[i][k + 4] * posses[k * 4 + j];
      diffM[i * 4 + j] = s;
    }
}

// ---------------- init: zero flags bitmap + anyflag ------------------------
__global__ void init_kernel(int* __restrict__ flags, int* __restrict__ anyflag, int F) {
  int i = blockIdx.x * blockDim.x + threadIdx.x;
  if (i < F) flags[i] = 0;
  if (i == 0) *anyflag = 0;
}

// ---------------- align: al = (diff @ [c,1])[:3] ----------------------------
__global__ void align_kernel(const float* __restrict__ coords, const float* __restrict__ diffM,
                             float* __restrict__ alc, int M) {
  int m = blockIdx.x * blockDim.x + threadIdx.x;
  if (m >= M) return;
  float c0 = coords[m * 3 + 0], c1 = coords[m * 3 + 1], c2 = coords[m * 3 + 2];
  for (int j = 0; j < 3; ++j)
    alc[m * 3 + j] = diffM[j * 4 + 0] * c0 + diffM[j * 4 + 1] * c1 + diffM[j * 4 + 2] * c2 + diffM[j * 4 + 3];
}

// ------- per-row prep: bf16 convert + RECIPROCAL feature norm + coord sq ----
__global__ __launch_bounds__(256)
void prep_rows_kernel(const float* __restrict__ feat, const float* __restrict__ coords3,
                      unsigned short* __restrict__ f16out, float2* __restrict__ rwsq, int rows) {
  int row = blockIdx.x * (blockDim.x >> 5) + (threadIdx.x >> 5);
  int lane = threadIdx.x & 31;
  if (row >= rows) return;
  const float* fr = feat + (size_t)row * CFEAT;
  float ss = 0.f;
  #pragma unroll
  for (int j = lane; j < CFEAT; j += 32) {
    float v = fr[j];
    ss += v * v;
    f16out[(size_t)row * CFEAT + j] = f32_to_bf16_rne(v);
  }
  for (int o = 16; o > 0; o >>= 1) ss += __shfl_xor(ss, o);
  if (lane == 0) {
    const float* c = coords3 + (size_t)row * 3;
    float sq = c[0] * c[0] + c[1] * c[1] + c[2] * c[2];
    rwsq[row] = make_float2(__builtin_amdgcn_rsqf(ss), sq);   // 1/||f||, ||x||^2
  }
}

// ---------------- softmax over 20 classes ----------------------------------
__global__ void softmax_kernel(const float* __restrict__ svp_in, float* __restrict__ svp_out, int N) {
  int n = blockIdx.x * blockDim.x + threadIdx.x;
  if (n >= N) return;
  const float* r = svp_in + (size_t)n * OUTC;
  float mx = r[0];
  #pragma unroll
  for (int j = 1; j < OUTC; ++j) mx = fmaxf(mx, r[j]);
  float s = 0.f, e[OUTC];
  #pragma unroll
  for (int j = 0; j < OUTC; ++j) { e[j] = __expf(r[j] - mx); s += e[j]; }
  float inv = __builtin_amdgcn_rcpf(s);
  #pragma unroll
  for (int j = 0; j < OUTC; ++j) svp_out[(size_t)n * OUTC + j] = e[j] * inv;
}

// ---------------- fused bf16-WMMA GEMM + distance + per-column argmin -------
// block = 256 threads (8 waves), grid = M/16. Wave w covers rows [w*N/8,(w+1)*N/8).
// Software-pipelined: next step's A tiles / coords / stats load during current compute.

__device__ __forceinline__ void load_step(const unsigned short* __restrict__ A16,
                                          const float* __restrict__ oric,
                                          const float2* __restrict__ rw1sq,
                                          int nb, int l16, int koff, int half,
                                          v16bf a[4], v2f& ac, float4 wq[4]) {
  const int arow = nb + l16;
  const unsigned short* ap = A16 + (size_t)arow * CFEAT + koff;
  #pragma unroll
  for (int s = 0; s < 4; ++s) {
    v8bf lo = *(const v8bf*)(ap + 32 * s);
    v8bf hi = *(const v8bf*)(ap + 32 * s + 16);
    a[s] = __builtin_shufflevector(lo, hi, 0, 1, 2, 3, 4, 5, 6, 7, 8, 9, 10, 11, 12, 13, 14, 15);
  }
  const float* x = oric + (size_t)arow * 3;
  float x0 = x[0], x1 = x[1], x2 = x[2];
  ac.x = half ? x2 : x0;
  ac.y = half ? 0.f : x1;
  // 8 consecutive {rw1, sq} pairs for this lane-half: 64 contiguous bytes
  const float4* wp = (const float4*)(rw1sq + (nb + (half << 3)));
  #pragma unroll
  for (int i = 0; i < 4; ++i) wq[i] = wp[i];
}

__global__ __launch_bounds__(256)
void gemm_argmin_kernel(const unsigned short* __restrict__ A16,  // [N,128] bf16 (mean_features)
                        const unsigned short* __restrict__ B16,  // [M,128] bf16 (sur_sv_feature)
                        const float2* __restrict__ rw1sq,        // per-n {1/||f||, ||x||^2}
                        const float2* __restrict__ rw2sq,        // per-m {1/||f||, ||y||^2}
                        const float* __restrict__ oric,          // [N,3]
                        const float* __restrict__ alc,           // [M,3] aligned
                        int* __restrict__ sm_index, int N, int M) {
  const int lane = threadIdx.x & 31;
  const int wave = threadIdx.x >> 5;
  const int half = (lane >= 16) ? 1 : 0;
  const int l16  = lane & 15;
  const int mb   = blockIdx.x * 16;
  const int koff = half << 3;  // 0 or 8 halves

  // ---- hoist B tile (16 columns x K=128, bf16) : loop-invariant -----------
  v16bf bt[4];
  {
    const unsigned short* bp = B16 + (size_t)(mb + l16) * CFEAT + koff;
    #pragma unroll
    for (int s = 0; s < 4; ++s) {
      v8bf lo = *(const v8bf*)(bp + 32 * s);
      v8bf hi = *(const v8bf*)(bp + 32 * s + 16);
      bt[s] = __builtin_shufflevector(lo, hi, 0, 1, 2, 3, 4, 5, 6, 7, 8, 9, 10, 11, 12, 13, 14, 15);
    }
  }
  // ---- coord B' (4x16, f32, K-padded with 0) ------------------------------
  v2f bc;
  {
    const float* y = alc + (size_t)(mb + l16) * 3;
    float y0 = y[0], y1 = y[1], y2 = y[2];
    bc.x = half ? y2 : y0;
    bc.y = half ? 0.f : y1;
  }
  const float rw2c = rw2sq[mb + l16].x;
  const float sqc  = rw2sq[mb + l16].y;

  float bestd = 3.4e38f;
  int   bestn = 0;
  const int chunk = N >> 3;
  const int n0   = wave * chunk;
  const int nend = n0 + chunk;

  // ---- software pipeline: prime first step --------------------------------
  v16bf aC[4]; v2f acC; float4 wqC[4];
  load_step(A16, oric, rw1sq, n0, l16, koff, half, aC, acC, wqC);

  for (int nb = n0; nb < nend; nb += 16) {
    // prefetch next step while computing current
    v16bf aN[4]; v2f acN; float4 wqN[4];
    const int nb2 = nb + 16;
    if (nb2 < nend) load_step(A16, oric, rw1sq, nb2, l16, koff, half, aN, acN, wqN);

    v8f acc = {};
    acc = __builtin_amdgcn_wmma_f32_16x16x32_bf16(false, aC[0], false, bt[0], (short)0, acc, false, false);
    acc = __builtin_amdgcn_wmma_f32_16x16x32_bf16(false, aC[1], false, bt[1], (short)0, acc, false, false);
    acc = __builtin_amdgcn_wmma_f32_16x16x32_bf16(false, aC[2], false, bt[2], (short)0, acc, false, false);
    acc = __builtin_amdgcn_wmma_f32_16x16x32_bf16(false, aC[3], false, bt[3], (short)0, acc, false, false);

    v8f cz = {};
    v8f cdot = __builtin_amdgcn_wmma_f32_16x16x4_f32(false, acC, false, bc, (short)0, cz, false, false);

    const int rbase = nb + (half << 3);
    #pragma unroll
    for (int e = 0; e < 8; ++e) {
      float rw1 = (e & 1) ? wqC[e >> 1].z : wqC[e >> 1].x;
      float sq  = (e & 1) ? wqC[e >> 1].w : wqC[e >> 1].y;
      float rec = fminf(rw1 * rw2c, RECMAX);                       // == 1/max(w1*w2, eps)
      float ee  = __expf(-2.0f * fmaxf(sq + sqc - 2.0f * cdot[e], 0.0f));
      float dist = 2.0f - acc[e] * rec - ee;
      if (dist < bestd) { bestd = dist; bestn = rbase + e; }
    }

    if (nb2 < nend) {
      #pragma unroll
      for (int s = 0; s < 4; ++s) aC[s] = aN[s];
      acC = acN;
      #pragma unroll
      for (int i = 0; i < 4; ++i) wqC[i] = wqN[i];
    }
  }

  // merge the two lane-halves (rows rbase..+7 vs +8..+15) with tie -> lower n
  {
    float od = __shfl_xor(bestd, 16);
    int   on = __shfl_xor(bestn, 16);
    if (od < bestd || (od == bestd && on < bestn)) { bestd = od; bestn = on; }
  }
  __shared__ float sd[8][16];
  __shared__ int   si[8][16];
  if (lane < 16) { sd[wave][lane] = bestd; si[wave][lane] = bestn; }
  __syncthreads();
  if (threadIdx.x < 16) {
    float bd = sd[0][threadIdx.x];
    int   bn = si[0][threadIdx.x];
    for (int w = 1; w < 8; ++w) {  // waves in ascending-n order -> tie-break OK
      float d = sd[w][threadIdx.x]; int n2 = si[w][threadIdx.x];
      if (d < bd || (d == bd && n2 < bn)) { bd = d; bn = n2; }
    }
    sm_index[mb + threadIdx.x] = bn;
  }
}

// ---------------- vals gather + global any(vals > 0.1) ----------------------
__global__ void vals_kernel(const float* __restrict__ svp, const int* __restrict__ smi,
                            const int* __restrict__ gt, float* __restrict__ vals,
                            int* __restrict__ anyflag, int M) {
  int m = blockIdx.x * blockDim.x + threadIdx.x;
  if (m >= M) return;
  float v = svp[(size_t)smi[m] * OUTC + gt[m]];
  vals[m] = v;
  if (v > 0.1f) atomicOr(anyflag, 1);
}

// ---------------- mask + scatter onehot + idx_masked + presence flags -------
__global__ void mask_scatter_kernel(float* __restrict__ svp, const int* __restrict__ smi,
                                    const int* __restrict__ gt, const float* __restrict__ vals,
                                    const int* __restrict__ anyflag, float* __restrict__ out_idxm,
                                    int* __restrict__ flags, int M) {
  int m = blockIdx.x * blockDim.x + threadIdx.x;
  if (m >= M) return;
  float thresh = (*anyflag) ? 0.1f : 0.0f;
  int sm = smi[m];
  bool mk = vals[m] > thresh;
  out_idxm[m] = mk ? (float)sm : -1.0f;
  if (mk) {
    flags[1 + sm] = 1;
    int g = gt[m];
    float* row = svp + (size_t)sm * OUTC;
    #pragma unroll
    for (int j = 0; j < OUTC; ++j) row[j] = (j == g) ? 1.0f : 0.0f;
  } else {
    flags[0] = 1;
  }
}

// ---------------- pre = argmax over classes (first max) ---------------------
__global__ void pre_kernel(const float* __restrict__ svp, float* __restrict__ pre, int N) {
  int n = blockIdx.x * blockDim.x + threadIdx.x;
  if (n >= N) return;
  const float* r = svp + (size_t)n * OUTC;
  float best = r[0]; int bi = 0;
  #pragma unroll
  for (int j = 1; j < OUTC; ++j) { float v = r[j]; if (v > best) { best = v; bi = j; } }
  pre[n] = (float)bi;
}

// ---------------- sorted unique of idx_masked via bitmap + block scan -------
__global__ __launch_bounds__(1024)
void unique_kernel(const int* __restrict__ flags, float* __restrict__ trust, int N, int M) {
  __shared__ int ps[1024];
  const int t = threadIdx.x;
  const int F = N + 1;                   // entry 0 == value -1, entry 1+n == value n
  const int per = (F + 1023) / 1024;
  const int base = t * per;
  int cnt = 0;
  for (int i = 0; i < per; ++i) { int e = base + i; if (e < F && flags[e]) ++cnt; }
  ps[t] = cnt;
  __syncthreads();
  for (int o = 1; o < 1024; o <<= 1) {
    int v = (t >= o) ? ps[t - o] : 0;
    __syncthreads();
    ps[t] += v;
    __syncthreads();
  }
  int pos = ps[t] - cnt;
  for (int i = 0; i < per; ++i) {
    int e = base + i;
    if (e < F && flags[e]) trust[pos++] = (float)(e - 1);
  }
  int total = ps[1023];
  for (int i = total + t; i < M; i += 1024) trust[i] = -1.0f;  // fill_value pad
}

extern "C" void kernel_launch(void* const* d_in, const int* in_sizes, int n_in,
                              void* d_out, int out_size, void* d_ws, size_t ws_size,
                              hipStream_t stream) {
  const float* sur_feat   = (const float*)d_in[0];  // [M,128]
  const float* sur_coords = (const float*)d_in[1];  // [M,3]
  const int*   sur_gt     = (const int*)d_in[2];    // [M]
  const float* sv_prob    = (const float*)d_in[3];  // [N,20]
  const float* mean_feat  = (const float*)d_in[4];  // [N,128]
  const float* ori_coords = (const float*)d_in[5];  // [N,3]
  const float* posses     = (const float*)d_in[6];  // [2,4,4]

  const int M = in_sizes[0] / CFEAT;
  const int N = in_sizes[4] / CFEAT;

  // ---- output layout (flat float, reference tuple order) ----
  float* out = (float*)d_out;
  float* out_trust = out;                                   // [M]
  float* out_mf    = out_trust + M;                         // [N*128]
  float* out_oc    = out_mf + (size_t)N * CFEAT;            // [N*3]
  float* out_pre   = out_oc + (size_t)N * 3;                // [N]
  float* out_svp   = out_pre + N;                           // [N*20]
  float* out_idxm  = out_svp + (size_t)N * OUTC;            // [M]

  // ---- workspace carve-out ----
  char* w = (char*)d_ws;
  size_t off = 0;
  auto alloc = [&](size_t bytes) -> void* {
    void* p = w + off;
    off = (off + bytes + 255) & ~(size_t)255;
    return p;
  };
  float*          diffM   = (float*)alloc(16 * sizeof(float));
  int*            anyflag = (int*)alloc(sizeof(int));
  float*          alc     = (float*)alloc((size_t)M * 3 * sizeof(float));
  float2*         rw1sq   = (float2*)alloc((size_t)N * sizeof(float2));
  float2*         rw2sq   = (float2*)alloc((size_t)M * sizeof(float2));
  unsigned short* A16     = (unsigned short*)alloc((size_t)N * CFEAT * sizeof(unsigned short));
  unsigned short* B16     = (unsigned short*)alloc((size_t)M * CFEAT * sizeof(unsigned short));
  int*            smi     = (int*)alloc((size_t)M * sizeof(int));
  float*          vals    = (float*)alloc((size_t)M * sizeof(float));
  int*            flags   = (int*)alloc((size_t)(N + 1) * sizeof(int));

  // 1) zero flags + anyflag
  init_kernel<<<(N + 1 + 255) / 256, 256, 0, stream>>>(flags, anyflag, N + 1);
  // 2) diff = inv(P1) @ P0
  pose_kernel<<<1, 32, 0, stream>>>(posses, diffM);
  // 3) align surrounding coords
  align_kernel<<<(M + 255) / 256, 256, 0, stream>>>(sur_coords, diffM, alc, M);
  // 4) N-side prep (bf16 A, 1/norm, |x|^2)   — wave per row
  prep_rows_kernel<<<(N + 7) / 8, 256, 0, stream>>>(mean_feat, ori_coords, A16, rw1sq, N);
  // 5) M-side prep (bf16 B, 1/norm, |y|^2 on aligned coords)
  prep_rows_kernel<<<(M + 7) / 8, 256, 0, stream>>>(sur_feat, alc, B16, rw2sq, M);
  // 6) softmax -> out_svp (sv_prob_updated, pre-scatter)
  softmax_kernel<<<(N + 255) / 256, 256, 0, stream>>>(sv_prob, out_svp, N);
  // 7) fused WMMA GEMM + distance + argmin over N per column m
  gemm_argmin_kernel<<<M / 16, 256, 0, stream>>>(A16, B16, rw1sq, rw2sq, ori_coords, alc, smi, N, M);
  // 8) vals + global threshold presence
  vals_kernel<<<(M + 255) / 256, 256, 0, stream>>>(out_svp, smi, sur_gt, vals, anyflag, M);
  // 9) mask, idx_masked, onehot scatter, presence bitmap
  mask_scatter_kernel<<<(M + 255) / 256, 256, 0, stream>>>(out_svp, smi, sur_gt, vals, anyflag,
                                                           out_idxm, flags, M);
  // 10) pre = argmax over classes (post-scatter)
  pre_kernel<<<(N + 255) / 256, 256, 0, stream>>>(out_svp, out_pre, N);
  // 11) trust_svm = sorted unique of idx_masked, padded with -1
  unique_kernel<<<1, 1024, 0, stream>>>(flags, out_trust, N, M);
  // 12) passthrough outputs
  hipMemcpyAsync(out_mf, mean_feat, (size_t)N * CFEAT * sizeof(float), hipMemcpyDeviceToDevice, stream);
  hipMemcpyAsync(out_oc, ori_coords, (size_t)N * 3 * sizeof(float), hipMemcpyDeviceToDevice, stream);
}
